// ExpansionContrastModule_26886495273260
// MI455X (gfx1250) — matrix operand order repaired and checked by reading, not compiled, not run on hardware
//
#include <hip/hip_runtime.h>

#define H_ 128
#define W_ 128
#define HW 16384
#define CIN 32
#define BATCH 2
#define NH 4
#define HID 64

typedef _Float16 half_t;
typedef __attribute__((ext_vector_type(16))) _Float16 v16h;
typedef __attribute__((ext_vector_type(8)))  float    v8f;

// ---------------- WMMA fragment helpers (v_wmma_f32_16x16x32_f16) ----------------
// A (16x32, MxK): lane l<16 holds row M=l, K chunks {0..7} and {16..23};
//                 lane l>=16 holds row M=l-16, K chunks {8..15} and {24..31}.
__device__ __forceinline__ v16h frag_a_ld(const half_t* base, int stride, int row16, int lane) {
    int row = row16 + (lane & 15);
    int k0  = (lane >> 4) << 3;
    const half_t* p = base + row * stride + k0;
    union { v16h v; float4 q[2]; } u;
    u.q[0] = *(const float4*)(p);
    u.q[1] = *(const float4*)(p + 16);
    return u.v;
}
// B (32x16, KxN): lane n<16 holds col N=n, K chunks {0..7} and {16..23}; lane n>=16 holds col N=n-16, K chunks {8..15} and {24..31}.
__device__ __forceinline__ v16h frag_b_ld(const half_t* base, int stride, int col16, int lane) {
    int col = col16 + (lane & 15);
    int k0  = (lane >> 4) << 3;
    const half_t* p = base + col * stride + k0;
    union { v16h v; float4 q[2]; } u;
    u.q[0] = *(const float4*)(p);
    u.q[1] = *(const float4*)(p + 16);
    return u.v;
}
__device__ __forceinline__ v8f wmma_f16(v16h a, v16h b, v8f c) {
    return __builtin_amdgcn_wmma_f32_16x16x32_f16(false, a, false, b, (short)0, c, false, false);
}

// ---------------- Kernel 1: dilated 3x3 conv (zero pad) + bias + relu ----------------
__global__ void conv_h1_kernel(const float* __restrict__ cen,
                               const float* __restrict__ w1,   // [32][32][3][3]
                               const float* __restrict__ b1,   // [32]
                               float* __restrict__ h1, int d) {
    __shared__ float wsh[CIN * 9 * CIN];   // [ci][tap][o]
    int tid = threadIdx.x;
    for (int idx = tid; idx < CIN * 9 * CIN; idx += 256) {
        int ci = idx / (9 * CIN); int rem = idx % (9 * CIN);
        int tap = rem / CIN; int o = rem % CIN;
        wsh[idx] = w1[(o * CIN + ci) * 9 + tap];
    }
    __syncthreads();
    int p = blockIdx.x * 256 + tid;
    int b = p / HW, s = p % HW;
    int i0 = s / W_, j0 = s % W_;
    float acc[CIN];
    #pragma unroll
    for (int o = 0; o < CIN; ++o) acc[o] = b1[o];
    for (int ci = 0; ci < CIN; ++ci) {
        const float* cb = cen + ((size_t)(b * CIN + ci)) * HW;
        #pragma unroll
        for (int tap = 0; tap < 9; ++tap) {
            int rr = i0 + (tap / 3 - 1) * d;
            int cc = j0 + (tap % 3 - 1) * d;
            float x = 0.0f;
            if (rr >= 0 && rr < H_ && cc >= 0 && cc < W_) x = cb[rr * W_ + cc];
            const float* wp = &wsh[(ci * 9 + tap) * CIN];
            #pragma unroll
            for (int o = 0; o < CIN; ++o) acc[o] = fmaf(x, wp[o], acc[o]);
        }
    }
    #pragma unroll
    for (int o = 0; o < CIN; ++o)
        h1[((size_t)(b * CIN + o)) * HW + s] = fmaxf(acc[o], 0.0f);
}

// ---------------- Kernel 2: gates, sum_x/cen_x, sur (f16), Q projection ----------------
__global__ void prep_kernel(const float* __restrict__ cen,
                            const float* __restrict__ h1,
                            const float* __restrict__ sw2,   // [8][32]
                            const float* __restrict__ b2,    // [8]
                            const float* __restrict__ sumw,  // [2]
                            const float* __restrict__ qw,    // [64][32]
                            half_t* __restrict__ surBuf,     // [B][256][HW]
                            half_t* __restrict__ qBuf,       // [B][256][HW] row = 4o+shift
                            int d, int shift) {
    __shared__ float qwL[HID * CIN];
    __shared__ float sw2L[8 * CIN];
    __shared__ float b2L[8];
    int tid = threadIdx.x;
    for (int idx = tid; idx < HID * CIN; idx += 256) qwL[idx] = qw[idx];
    for (int idx = tid; idx < 8 * CIN; idx += 256) sw2L[idx] = sw2[idx];
    if (tid < 8) b2L[tid] = b2[tid];
    __syncthreads();
    float s0 = sumw[0], s1 = sumw[1];
    float wmx = fmaxf(s0, s1);
    float e0 = __expf(s0 - wmx), e1 = __expf(s1 - wmx);
    float w2a = e0 / (e0 + e1), w2b = e1 / (e0 + e1);
    int p = blockIdx.x * 256 + tid;
    int b = p / HW, s = p % HW;
    int i0 = s / W_, j0 = s % W_;
    float sw[8];
    {
        float h1v[CIN];
        #pragma unroll
        for (int c = 0; c < CIN; ++c) h1v[c] = h1[((size_t)(b * CIN + c)) * HW + s];
        float lm = -1e30f;
        #pragma unroll
        for (int o = 0; o < 8; ++o) {
            float a = b2L[o];
            #pragma unroll
            for (int c = 0; c < CIN; ++c) a = fmaf(sw2L[o * CIN + c], h1v[c], a);
            sw[o] = a; lm = fmaxf(lm, a);
        }
        float es = 0.0f;
        #pragma unroll
        for (int o = 0; o < 8; ++o) { sw[o] = __expf(sw[o] - lm); es += sw[o]; }
        float inv = w2a / es;
        #pragma unroll
        for (int o = 0; o < 8; ++o) sw[o] *= inv;
    }
    // reflected neighbor coordinates
    int rm = i0 - d; rm = rm < 0 ? -rm : rm;
    int rp = i0 + d; if (rp > H_ - 1) rp = 2 * (H_ - 1) - rp;
    int cm = j0 - d; cm = cm < 0 ? -cm : cm;
    int cp = j0 + d; if (cp > W_ - 1) cp = 2 * (W_ - 1) - cp;
    int nr[8] = {rm, rm, rm, i0, i0, rp, rp, rp};
    int nc[8] = {cm, j0, cp, cm, cp, cm, j0, cp};
    float sumx[CIN], cenx[CIN];
    #pragma unroll
    for (int c = 0; c < CIN; ++c) { sumx[c] = 0.0f; cenx[c] = 0.0f; }
    for (int k = 0; k < 8; ++k) {
        int off = nr[k] * W_ + nc[k];
        #pragma unroll
        for (int c = 0; c < CIN; ++c) {
            float v = cen[((size_t)(b * CIN + c)) * HW + off];
            sumx[c] = fmaf(v, sw[k], sumx[c]);
            cenx[c] += v;
        }
    }
    #pragma unroll
    for (int c = 0; c < CIN; ++c) {
        float cv = cen[((size_t)(b * CIN + c)) * HW + s];
        sumx[c] = fmaf(cv, w2b, sumx[c]);
        cenx[c] = cenx[c] * (w2a * 0.125f) + cv * w2b;
    }
    for (int k = 0; k < 8; ++k) {
        int off = nr[k] * W_ + nc[k];
        #pragma unroll
        for (int c = 0; c < CIN; ++c) {
            float v = cen[((size_t)(b * CIN + c)) * HW + off];
            surBuf[((size_t)(b * 8 + k) * CIN + c) * HW + s] = (half_t)(v - sumx[c]);
        }
    }
    // Q projection; stacked-head reshape collapses to row = 4o + shift
    size_t qbase = ((size_t)(b * 256 + shift)) * HW + s;
    for (int o = 0; o < HID; ++o) {
        float a = 0.0f;
        #pragma unroll
        for (int c = 0; c < CIN; ++c) a = fmaf(qwL[o * CIN + c], cenx[c], a);
        qBuf[qbase + (size_t)(4 * o) * HW] = (half_t)a;
    }
}

// ---------------- Kernel 3: generic WMMA GEMM, 128x128 block tile, K-step 32 ----------
// MODE 0: K/V projection, f16 out, row = b*2048 + 4*c_out + shift (head interleave).
// MODE 1: final projection, f32 out, row = b*M + c_out.
template <int MODE>
__global__ void gemm128_kernel(const float* __restrict__ Wg,   // [M][Kd] f32
                               const half_t* __restrict__ X,   // [B][Kd][HW] f16
                               half_t* __restrict__ dstH,
                               float* __restrict__ dstF,
                               int M, int Kd, int shift) {
    __shared__ half_t As[128 * 40];
    __shared__ half_t Bs[128 * 40];
    int tid = threadIdx.x, lane = tid & 31, wv = tid >> 5;
    int b = blockIdx.z;
    int mBlock = blockIdx.y * 128;
    int nBlock = blockIdx.x * 128;
    const half_t* Xb = X + (size_t)b * Kd * HW;
    int mSub = (wv & 3) * 32;
    int nSub = (wv >> 2) * 64;
    // tile-loader coordinates (hoisted)
    int am = tid >> 1, akh = (tid & 1) * 16;
    int bc = tid >> 3, bsL = (tid & 7) * 16;
    const float*  aSrc0 = Wg + (size_t)(mBlock + am) * Kd + akh;
    const half_t* bSrc0 = Xb + (size_t)bc * HW + nBlock + bsL;
    v8f acc[2][4];
    for (int i = 0; i < 2; ++i) for (int j = 0; j < 4; ++j)
        for (int e = 0; e < 8; ++e) acc[i][j][e] = 0.0f;
    for (int k0 = 0; k0 < Kd; k0 += 32) {
        {   // A tile: cvt f32 weights -> f16 LDS [m][k]
            const float* src = aSrc0 + k0;
            half_t* dst = &As[am * 40 + akh];
            #pragma unroll
            for (int j = 0; j < 16; ++j) dst[j] = (half_t)src[j];
        }
        {   // B tile: X[k][s] -> LDS transposed Bs[s][k]
            const half_t* src = bSrc0 + (size_t)k0 * HW;
            union { float4 f; half_t h[8]; } u0, u1;
            u0.f = *(const float4*)(src);
            u1.f = *(const float4*)(src + 8);
            #pragma unroll
            for (int j = 0; j < 8; ++j) Bs[(bsL + j) * 40 + bc] = u0.h[j];
            #pragma unroll
            for (int j = 0; j < 8; ++j) Bs[(bsL + 8 + j) * 40 + bc] = u1.h[j];
            if (k0 + 32 < Kd) {   // prefetch next B tile -> global_prefetch_b8
                __builtin_prefetch(bSrc0 + (size_t)(k0 + 32) * HW, 0, 1);
            }
        }
        __syncthreads();
        v16h af[2], bf[4];
        #pragma unroll
        for (int i = 0; i < 2; ++i) af[i] = frag_a_ld(As, 40, mSub + i * 16, lane);
        #pragma unroll
        for (int j = 0; j < 4; ++j) bf[j] = frag_b_ld(Bs, 40, nSub + j * 16, lane);
        #pragma unroll
        for (int i = 0; i < 2; ++i)
            #pragma unroll
            for (int j = 0; j < 4; ++j)
                acc[i][j] = wmma_f16(af[i], bf[j], acc[i][j]);
        __syncthreads();
    }
    int nL = lane & 15;
    int m0 = (lane >> 4) << 3;
    for (int i = 0; i < 2; ++i) for (int j = 0; j < 4; ++j) {
        int n = nBlock + nSub + j * 16 + nL;
        int mBase = mBlock + mSub + i * 16 + m0;
        if (MODE == 0) {
            size_t base = ((size_t)(b * 2048 + 4 * mBase + shift)) * HW + n;
            #pragma unroll
            for (int r = 0; r < 8; ++r)
                dstH[base + (size_t)(4 * r) * HW] = (half_t)acc[i][j][r];
        } else {
            size_t base = ((size_t)(b * M + mBase)) * HW + n;
            #pragma unroll
            for (int r = 0; r < 8; ++r)
                dstF[base + (size_t)r * HW] = acc[i][j][r];
        }
    }
}

// ---------------- Kernel 4: per-row 1/max(||row||,1e-12) ----------------
__global__ void rownorm_kernel(const half_t* __restrict__ X, float* __restrict__ inv, int len) {
    __shared__ float red[256];
    int row = blockIdx.x, tid = threadIdx.x;
    const half_t* p = X + (size_t)row * len;
    float s = 0.0f;
    for (int idx = tid; idx < len; idx += 256) { float v = (float)p[idx]; s = fmaf(v, v, s); }
    red[tid] = s; __syncthreads();
    for (int st = 128; st > 0; st >>= 1) { if (tid < st) red[tid] += red[tid + st]; __syncthreads(); }
    if (tid == 0) inv[row] = 1.0f / fmaxf(sqrtf(red[0]), 1e-12f);
}

// ---------------- Kernel 5a: score = (Qn . Kn^T) * invq*invk/128, WMMA over s ---------
__global__ void score_kernel(const half_t* __restrict__ Q,   // [B*4][64][HW]
                             const half_t* __restrict__ K,   // [B*4][512][HW]
                             const float* __restrict__ qinv,
                             const float* __restrict__ kinv,
                             float* __restrict__ S) {        // [B*4][64][512]
    __shared__ half_t Qs[64 * 40];
    __shared__ half_t Ks[64 * 40];
    int tid = threadIdx.x, lane = tid & 31, wv = tid >> 5;
    int bn = blockIdx.y;
    int nBlock = blockIdx.x * 64;
    const half_t* Qb = Q + (size_t)bn * 64 * HW;
    const half_t* Kb = K + (size_t)bn * 512 * HW;
    int mSub = (wv & 3) * 16;
    int nSub = (wv >> 2) * 32;
    int r = tid >> 2, kh = (tid & 3) * 8;
    const half_t* qSrc0 = Qb + (size_t)r * HW + kh;
    const half_t* kSrc0 = Kb + (size_t)(nBlock + r) * HW + kh;
    v8f acc[2];
    for (int j = 0; j < 2; ++j) for (int e = 0; e < 8; ++e) acc[j][e] = 0.0f;
    for (int s0 = 0; s0 < HW; s0 += 32) {
        *(float4*)&Qs[r * 40 + kh] = *(const float4*)(qSrc0 + s0);
        *(float4*)&Ks[r * 40 + kh] = *(const float4*)(kSrc0 + s0);
        if (s0 + 32 < HW) {
            __builtin_prefetch(qSrc0 + s0 + 32, 0, 1);
            __builtin_prefetch(kSrc0 + s0 + 32, 0, 1);
        }
        __syncthreads();
        v16h a = frag_a_ld(Qs, 40, mSub, lane);
        #pragma unroll
        for (int j = 0; j < 2; ++j) {
            v16h bf = frag_b_ld(Ks, 40, nSub + j * 16, lane);
            acc[j] = wmma_f16(a, bf, acc[j]);
        }
        __syncthreads();
    }
    int nL = lane & 15, m0 = (lane >> 4) << 3;
    for (int j = 0; j < 2; ++j) {
        int kcol = nBlock + nSub + j * 16 + nL;
        #pragma unroll
        for (int rr = 0; rr < 8; ++rr) {
            int q = mSub + m0 + rr;
            S[((size_t)bn * 64 + q) * 512 + kcol] =
                acc[j][rr] * qinv[bn * 64 + q] * kinv[bn * 512 + kcol] * 0.0078125f;
        }
    }
}

// ---------------- Kernel 5b: instance-norm over 64x512 then row softmax ---------------
__global__ void inorm_softmax_kernel(const float* __restrict__ S, half_t* __restrict__ A) {
    __shared__ float red[256];
    int bn = blockIdx.x, tid = threadIdx.x;
    const float* Sb = S + (size_t)bn * 64 * 512;
    half_t* Ab = A + (size_t)bn * 64 * 512;
    float s1 = 0.0f, s2 = 0.0f;
    for (int idx = tid; idx < 64 * 512; idx += 256) {
        float v = Sb[idx]; s1 += v; s2 = fmaf(v, v, s2);
    }
    red[tid] = s1; __syncthreads();
    for (int st = 128; st > 0; st >>= 1) { if (tid < st) red[tid] += red[tid + st]; __syncthreads(); }
    float mean = red[0] * (1.0f / 32768.0f); __syncthreads();
    red[tid] = s2; __syncthreads();
    for (int st = 128; st > 0; st >>= 1) { if (tid < st) red[tid] += red[tid + st]; __syncthreads(); }
    float var = red[0] * (1.0f / 32768.0f) - mean * mean; __syncthreads();
    float istd = rsqrtf(var + 1e-5f);
    for (int r = 0; r < 64; ++r) {
        float v0 = (Sb[r * 512 + tid] - mean) * istd;
        float v1 = (Sb[r * 512 + 256 + tid] - mean) * istd;
        red[tid] = fmaxf(v0, v1); __syncthreads();
        for (int st = 128; st > 0; st >>= 1) { if (tid < st) red[tid] = fmaxf(red[tid], red[tid + st]); __syncthreads(); }
        float rowmax = red[0]; __syncthreads();
        float e0 = __expf(v0 - rowmax), e1 = __expf(v1 - rowmax);
        red[tid] = e0 + e1; __syncthreads();
        for (int st = 128; st > 0; st >>= 1) { if (tid < st) red[tid] += red[tid + st]; __syncthreads(); }
        float rs = 1.0f / red[0]; __syncthreads();
        Ab[r * 512 + tid] = (half_t)(e0 * rs);
        Ab[r * 512 + 256 + tid] = (half_t)(e1 * rs);
    }
}

// ---------------- Kernel 5c: out = attn @ V, WMMA, M=64, K=512, N=16384 ---------------
__global__ void outgemm_kernel(const half_t* __restrict__ A,   // [B*4][64][512]
                               const half_t* __restrict__ V,   // [B*4][512][HW]
                               half_t* __restrict__ O) {       // [B*4][64][HW]
    __shared__ half_t As[64 * 40];
    __shared__ half_t Bs[128 * 40];
    int tid = threadIdx.x, lane = tid & 31, wv = tid >> 5;
    int bn = blockIdx.y;
    int nBlock = blockIdx.x * 128;
    const half_t* Ab = A + (size_t)bn * 64 * 512;
    const half_t* Vb = V + (size_t)bn * 512 * HW;
    int mSub = (wv & 3) * 16;
    int nSub = (wv >> 2) * 64;
    int ar = tid >> 2, akh = (tid & 3) * 8;
    int bc = tid >> 3, bsL = (tid & 7) * 16;
    const half_t* aSrc0 = Ab + ar * 512 + akh;
    const half_t* bSrc0 = Vb + (size_t)bc * HW + nBlock + bsL;
    v8f acc[4];
    for (int j = 0; j < 4; ++j) for (int e = 0; e < 8; ++e) acc[j][e] = 0.0f;
    for (int k0 = 0; k0 < 512; k0 += 32) {
        *(float4*)&As[ar * 40 + akh] = *(const float4*)(aSrc0 + k0);
        {
            const half_t* src = bSrc0 + (size_t)k0 * HW;
            union { float4 f; half_t h[8]; } u0, u1;
            u0.f = *(const float4*)(src);
            u1.f = *(const float4*)(src + 8);
            #pragma unroll
            for (int j = 0; j < 8; ++j) Bs[(bsL + j) * 40 + bc] = u0.h[j];
            #pragma unroll
            for (int j = 0; j < 8; ++j) Bs[(bsL + 8 + j) * 40 + bc] = u1.h[j];
            if (k0 + 32 < 512) __builtin_prefetch(bSrc0 + (size_t)(k0 + 32) * HW, 0, 1);
        }
        __syncthreads();
        v16h a = frag_a_ld(As, 40, mSub, lane);
        #pragma unroll
        for (int j = 0; j < 4; ++j) {
            v16h bf = frag_b_ld(Bs, 40, nSub + j * 16, lane);
            acc[j] = wmma_f16(a, bf, acc[j]);
        }
        __syncthreads();
    }
    int nL = lane & 15, m0 = (lane >> 4) << 3;
    for (int j = 0; j < 4; ++j) {
        int s = nBlock + nSub + j * 16 + nL;
        size_t base = ((size_t)bn * 64 + mSub + m0) * HW + s;
        #pragma unroll
        for (int r = 0; r < 8; ++r)
            O[base + (size_t)r * HW] = (half_t)acc[j][r];
    }
}

// ---------------- Kernels 7/8: batch-norm (batch stats) + relu ----------------
__global__ void bnstats_kernel(const float* __restrict__ Y, float* __restrict__ mean,
                               float* __restrict__ istd) {
    __shared__ float red[256];
    int o = blockIdx.x, tid = threadIdx.x;
    float s1 = 0.0f, s2 = 0.0f;
    for (int b = 0; b < BATCH; ++b) {
        const float* p = Y + ((size_t)b * 256 + o) * HW;
        for (int idx = tid; idx < HW; idx += 256) { float v = p[idx]; s1 += v; s2 = fmaf(v, v, s2); }
    }
    red[tid] = s1; __syncthreads();
    for (int st = 128; st > 0; st >>= 1) { if (tid < st) red[tid] += red[tid + st]; __syncthreads(); }
    float m = red[0] * (1.0f / (BATCH * HW)); __syncthreads();
    red[tid] = s2; __syncthreads();
    for (int st = 128; st > 0; st >>= 1) { if (tid < st) red[tid] += red[tid + st]; __syncthreads(); }
    if (tid == 0) {
        float var = red[0] * (1.0f / (BATCH * HW)) - m * m;
        mean[o] = m; istd[o] = rsqrtf(var + 1e-5f);
    }
}

__global__ void bnapply_kernel(const float* __restrict__ Y, const float* __restrict__ mean,
                               const float* __restrict__ istd, const float* __restrict__ gamma,
                               const float* __restrict__ beta, float* __restrict__ out) {
    size_t idx = (size_t)blockIdx.x * 256 + threadIdx.x;
    int o = (int)((idx / HW) % 256);
    float v = (Y[idx] - mean[o]) * istd[o] * gamma[o] + beta[o];
    out[idx] = fmaxf(v, 0.0f);
}

// ---------------- Host-side orchestration ----------------
extern "C" void kernel_launch(void* const* d_in, const int* in_sizes, int n_in,
                              void* d_out, int out_size, void* d_ws, size_t ws_size,
                              hipStream_t stream) {
    const float* cen   = (const float*)d_in[0];
    const float* q_w   = (const float*)d_in[1];
    const float* k_w   = (const float*)d_in[2];
    const float* v_w   = (const float*)d_in[3];
    const float* sw1   = (const float*)d_in[4];
    const float* sb1   = (const float*)d_in[5];
    const float* sw2   = (const float*)d_in[6];
    const float* sb2   = (const float*)d_in[7];
    const float* sumw  = (const float*)d_in[8];
    const float* outw  = (const float*)d_in[9];
    const float* gamma = (const float*)d_in[10];
    const float* beta  = (const float*)d_in[11];
    float* out = (float*)d_out;
    (void)in_sizes; (void)n_in; (void)out_size; (void)ws_size;

    char* ws = (char*)d_ws;
    size_t off = 0;
    auto alloc = [&](size_t bytes) -> void* {
        void* p = ws + off;
        off = (off + bytes + 255) & ~(size_t)255;
        return p;
    };
    float*  h1    = (float*) alloc((size_t)BATCH * CIN * HW * 4);      //   4 MB
    half_t* sur   = (half_t*)alloc((size_t)BATCH * 256 * HW * 2);      //  16 MB
    half_t* Qbuf  = (half_t*)alloc((size_t)BATCH * 256 * HW * 2);      //  16 MB
    half_t* Kbuf  = (half_t*)alloc((size_t)BATCH * 2048 * HW * 2);     // 128 MB
    half_t* Vbuf  = (half_t*)alloc((size_t)BATCH * 2048 * HW * 2);     // 128 MB
    float*  qinv  = (float*) alloc((size_t)BATCH * 256 * 4);
    float*  kinv  = (float*) alloc((size_t)BATCH * 2048 * 4);
    float*  score = (float*) alloc((size_t)BATCH * 4 * 64 * 512 * 4);  //   1 MB
    half_t* attn  = (half_t*)alloc((size_t)BATCH * 4 * 64 * 512 * 2);
    half_t* obuf  = (half_t*)alloc((size_t)BATCH * 256 * HW * 2);      //  16 MB
    float*  ybuf  = (float*) alloc((size_t)BATCH * 256 * HW * 4);      //  32 MB
    float*  bnm   = (float*) alloc(256 * 4);
    float*  bns   = (float*) alloc(256 * 4);

    const int shifts[4] = {1, 2, 4, 8};
    for (int i = 0; i < 4; ++i) {
        int d = shifts[i];
        conv_h1_kernel<<<BATCH * HW / 256, 256, 0, stream>>>(
            cen, sw1 + (size_t)i * CIN * CIN * 9, sb1 + i * CIN, h1, d);
        prep_kernel<<<BATCH * HW / 256, 256, 0, stream>>>(
            cen, h1, sw2 + i * 8 * CIN, sb2 + i * 8, sumw + i * 2,
            q_w + (size_t)i * HID * CIN, sur, Qbuf, d, i);
        gemm128_kernel<0><<<dim3(HW / 128, 512 / 128, BATCH), 256, 0, stream>>>(
            k_w + (size_t)i * 512 * 256, sur, Kbuf, nullptr, 512, 256, i);
        gemm128_kernel<0><<<dim3(HW / 128, 512 / 128, BATCH), 256, 0, stream>>>(
            v_w + (size_t)i * 512 * 256, sur, Vbuf, nullptr, 512, 256, i);
    }
    rownorm_kernel<<<BATCH * 2048, 256, 0, stream>>>(Kbuf, kinv, HW);
    rownorm_kernel<<<BATCH * 256, 256, 0, stream>>>(Qbuf, qinv, HW);
    score_kernel<<<dim3(512 / 64, BATCH * NH), 256, 0, stream>>>(Qbuf, Kbuf, qinv, kinv, score);
    inorm_softmax_kernel<<<BATCH * NH, 256, 0, stream>>>(score, attn);
    outgemm_kernel<<<dim3(HW / 128, BATCH * NH), 256, 0, stream>>>(attn, Vbuf, obuf);
    gemm128_kernel<1><<<dim3(HW / 128, 256 / 128, BATCH), 256, 0, stream>>>(
        outw, obuf, nullptr, ybuf, 256, 256, 0);
    bnstats_kernel<<<256, 256, 0, stream>>>(ybuf, bnm, bns);
    bnapply_kernel<<<(unsigned)((size_t)BATCH * 256 * HW / 256), 256, 0, stream>>>(
        ybuf, bnm, bns, gamma, beta, out);
}
